// CrossAttention_58789512347762
// MI455X (gfx1250) — matrix operand order repaired and checked by reading, not compiled, not run on hardware
//
#include <hip/hip_runtime.h>

// ---------------------------------------------------------------------------
// CDNA5 (gfx1250) cross-attention, wave32 + v_wmma_f32_16x16x32_f16 throughout
// Attention computed in the transposed domain (S^T = K Q^T, O^T = V P^T) so
// softmax stats are per-lane scalars and the P relayout is two packed
// ds_store_b128 + one 32B LDS vector load per 32-key step. Fragment loads are
// double-buffered in registers so WMMAs overlap outstanding VMEM.
// ---------------------------------------------------------------------------

typedef __attribute__((ext_vector_type(16))) _Float16 v16h;
typedef __attribute__((ext_vector_type(8)))  _Float16 v8h;
typedef __attribute__((ext_vector_type(8)))  float    v8f;

#define HEADS  8
#define DHEAD  64
#define QDIM   256
#define CDIM   512
#define HID    512
#define NQ     4096   // 64*64 query pixels
#define NK     1024   // 32*32 context pixels
#define NB     4
#define PSTR   48     // LDS row stride (halfs) for P^T tile: 96B, 32B-aligned rows

__device__ __forceinline__ v8f wmma32(v16h a, v16h b, v8f c) {
    // D = A(16x32 f16) * B(32x16 f16) + C(16x16 f32)
    return __builtin_amdgcn_wmma_f32_16x16x32_f16(
        /*neg_a=*/false, a, /*neg_b=*/false, b,
        /*c_mod=*/(short)0, c, /*reuse_a=*/false, /*reuse_b=*/false);
}

// A fragment (16x32, M x K) from row-major f32 [M][K], row stride ldk.
// Contiguous k order (same as the B loaders): lane L holds row M=L&15; halfs 0..15 -> K = hw+0..15,
// hw = 16*(L>=16).
__device__ __forceinline__ v16h load_a_frag_f32(const float* __restrict__ base,
                                                int ldk, int lane, int k0) {
    const int m  = lane & 15;
    const int hw = (lane >> 4) << 4;
    const float* r = base + (size_t)m * ldk + k0;
    v16h a;
#pragma unroll
    for (int e = 0; e < 8; ++e) {
        a[e]     = (_Float16)r[hw + e];
        a[e + 8] = (_Float16)r[hw + 8 + e];
    }
    return a;
}

// A fragment (16x32) from row-major f16 [M][K] — two contiguous 16B loads.
__device__ __forceinline__ v16h load_a_frag_f16(const _Float16* __restrict__ base,
                                                int ldk, int lane, int k0) {
    const int m  = lane & 15;
    const int hw = (lane >> 4) << 4;
    const _Float16* r = base + (size_t)m * ldk + k0;
    v8h lo = *(const v8h*)(r + hw);
    v8h hi = *(const v8h*)(r + hw + 8);
    v16h a;
#pragma unroll
    for (int e = 0; e < 8; ++e) { a[e] = lo[e]; a[e + 8] = hi[e]; }
    return a;
}

// B fragment (32x16, K x N) from K-major storage [K][N], row stride ldn.
template <typename T>
__device__ __forceinline__ v16h load_b_frag_kmajor(const T* __restrict__ base,
                                                   int ldn, int lane, int k0) {
    const int n  = lane & 15;
    const int kk = k0 + ((lane >> 4) << 4);
    const T* p = base + (size_t)kk * ldn + n;
    v16h b;
#pragma unroll
    for (int e = 0; e < 16; ++e) b[e] = (_Float16)p[(size_t)e * ldn];
    return b;
}

// B fragment (32x16) from N-major f16 storage [N][K] (K contiguous):
// one contiguous 32B load per lane.
__device__ __forceinline__ v16h load_b_frag_nmajor(const _Float16* __restrict__ base,
                                                   int ldk, int lane, int k0) {
    const int n  = lane & 15;
    const int kk = k0 + ((lane >> 4) << 4);
    return *(const v16h*)(base + (size_t)n * ldk + kk);
}

// ---------------------------------------------------------------------------
// Projection GEMM: Out[b][o][p] = sum_c W[o][c] * In[b][c][p]
// mode 0: f16 out, plain [o][p]               (V -> [d][j] per head)
// mode 1: f16 out, per-head transposed [p][d] (Q -> [p][d], K -> [j][d])
// mode 2: f32 out + bias, plain [o][p]        (final O projection)
// ---------------------------------------------------------------------------
template <typename TIn>
__global__ __launch_bounds__(256) void proj_kernel(
    const float* __restrict__ W, const TIn* __restrict__ In,
    _Float16* __restrict__ out16, float* __restrict__ out32,
    const float* __restrict__ bias, int M, int N, int K, int mode) {
    const int lane = threadIdx.x & 31;
    const int wid  = threadIdx.x >> 5;
    const int b    = blockIdx.y;
    const int ntN  = N >> 4;
    const int tile = blockIdx.x * 8 + wid;
    if (tile >= (M >> 4) * ntN) return;     // wave-uniform
    const int M0 = (tile / ntN) << 4;
    const int N0 = (tile % ntN) << 4;

    const TIn* inB = In + (size_t)b * K * N + N0;
    const float* wA = W + (size_t)M0 * K;

    v8f c = {};
    for (int k0 = 0; k0 < K; k0 += 32) {
        v16h a  = load_a_frag_f32(wA, K, lane, k0);
        v16h bb = load_b_frag_kmajor<TIn>(inB, N, lane, k0);
        c = wmma32(a, bb, c);
    }

    const int n    = N0 + (lane & 15);
    const int half = lane >> 4;
    if (mode == 2) {
#pragma unroll
        for (int r = 0; r < 8; ++r) {
            const int o = M0 + r + (half << 3);
            out32[(size_t)b * M * N + (size_t)o * N + n] = c[r] + bias[o];
        }
    } else if (mode == 1) {
#pragma unroll
        for (int r = 0; r < 8; ++r) {
            const int o = M0 + r + (half << 3);
            const int h = o >> 6, d = o & 63;
            out16[((size_t)(b * HEADS + h) * N + n) * DHEAD + d] = (_Float16)c[r];
        }
    } else {
#pragma unroll
        for (int r = 0; r < 8; ++r) {
            const int o = M0 + r + (half << 3);
            out16[(size_t)b * M * N + (size_t)o * N + n] = (_Float16)c[r];
        }
    }
}

// ---------------------------------------------------------------------------
// Flash attention, transposed domain, register-double-buffered fragments.
//   q_t : [b*H][4096][64]  (p rows, d contiguous)  -> B fragments (Q^T n-major)
//   k_t : [b*H][1024][64]  (j rows, d contiguous)  -> A fragments (S^T = K Q^T)
//   v_m : [b*H][64][1024]  ([d][j], j contiguous)  -> A fragments (O^T = V P^T)
//   o_m : [b][512][4096]   f16, feeds final projection
// ---------------------------------------------------------------------------
__global__ __launch_bounds__(256) void attn_kernel(
    const _Float16* __restrict__ q_t, const _Float16* __restrict__ k_t,
    const _Float16* __restrict__ v_m, _Float16* __restrict__ o_m) {
    __shared__ __align__(64) _Float16 smem[8][16 * PSTR];

    const int lane = threadIdx.x & 31;
    const int wid  = threadIdx.x >> 5;
    const int h    = blockIdx.y;
    const int b    = blockIdx.z;
    const int p0   = (blockIdx.x * 8 + wid) << 4;
    const size_t bh = (size_t)(b * HEADS + h);

    const _Float16* q  = q_t + bh * NQ * DHEAD + (size_t)p0 * DHEAD;
    const _Float16* kk = k_t + bh * NK * DHEAD;
    const _Float16* vv = v_m + bh * DHEAD * NK;

    // Q^T B-fragments, resident for the whole key loop
    const v16h bq0 = load_b_frag_nmajor(q, DHEAD, lane, 0);
    const v16h bq1 = load_b_frag_nmajor(q, DHEAD, lane, 32);

    float m = -1e30f, l = 0.f;
    v8f acc[4] = {{}, {}, {}, {}};   // O^T tiles: rows d = t*16 + r + 8*half

    _Float16* sp = smem[wid];
    const float scale = 0.125f;      // 64^-0.5
    const int half = lane >> 4;
    const int i    = lane & 15;      // query column owned by this lane

    // ---- prologue: fragments for j0 = 0
    v16h ak[4], av[4];
    ak[0] = load_a_frag_f16(kk,              DHEAD, lane, 0);
    ak[1] = load_a_frag_f16(kk,              DHEAD, lane, 32);
    ak[2] = load_a_frag_f16(kk + 16 * DHEAD, DHEAD, lane, 0);
    ak[3] = load_a_frag_f16(kk + 16 * DHEAD, DHEAD, lane, 32);
#pragma unroll
    for (int t = 0; t < 4; ++t)
        av[t] = load_a_frag_f16(vv + (size_t)(t << 4) * NK, NK, lane, 0);

    for (int j0 = 0; j0 < NK; j0 += 32) {
        const int jn = (j0 + 32) & (NK - 1);     // wraps to 0 on last step
        __builtin_prefetch(kk + (size_t)jn * DHEAD, 0, 1);

        // ---- S^T = K Q^T, two 16-key tiles (fragments already resident)
        v8f s0 = {}, s1 = {};
        s0 = wmma32(ak[0], bq0, s0);
        s0 = wmma32(ak[1], bq1, s0);
        s1 = wmma32(ak[2], bq0, s1);
        s1 = wmma32(ak[3], bq1, s1);

        // ---- start next K fragments while softmax runs
        const _Float16* knb = kk + (size_t)jn * DHEAD;
        ak[0] = load_a_frag_f16(knb,              DHEAD, lane, 0);
        ak[1] = load_a_frag_f16(knb,              DHEAD, lane, 32);
        ak[2] = load_a_frag_f16(knb + 16 * DHEAD, DHEAD, lane, 0);
        ak[3] = load_a_frag_f16(knb + 16 * DHEAD, DHEAD, lane, 32);

        // ---- online softmax, per-lane over this lane's 16 key values
        float mx = -1e30f;
#pragma unroll
        for (int r = 0; r < 8; ++r) {
            s0[r] *= scale; s1[r] *= scale;
            mx = fmaxf(mx, fmaxf(s0[r], s1[r]));
        }
        mx = fmaxf(mx, __shfl_xor(mx, 16, 32));   // combine the two key halves
        const float mn    = fmaxf(m, mx);
        const float alpha = __expf(m - mn);
        m = mn;

        v8h pk0, pk1;
        float rs = 0.f;
#pragma unroll
        for (int r = 0; r < 8; ++r) {
            const float e0 = __expf(s0[r] - mn);
            const float e1 = __expf(s1[r] - mn);
            rs += e0 + e1;
            pk0[r] = (_Float16)e0;
            pk1[r] = (_Float16)e1;
        }
        rs += __shfl_xor(rs, 16, 32);
        l = l * alpha + rs;

        // ---- P^T -> LDS: lane's 8 values are contiguous in j (packed stores)
        *(v8h*)(sp + i * PSTR + (half << 3))      = pk0;   // j = 8*half + r
        *(v8h*)(sp + i * PSTR + 16 + (half << 3)) = pk1;   // j = 16 + 8*half + r
        asm volatile("s_wait_dscnt 0x0" ::: "memory");
        // B fragment of P^T: 32 contiguous bytes along j
        const v16h bp = *(const v16h*)(sp + i * PSTR + (half << 4));

        // ---- O^T = alpha*O^T + V P^T  (4 tiles of 16 head-dims)
#pragma unroll
        for (int t = 0; t < 4; ++t) {
#pragma unroll
            for (int r = 0; r < 8; ++r) acc[t][r] *= alpha;
            acc[t] = wmma32(av[t], bp, acc[t]);
        }

        // ---- start next V fragments while acc WMMAs drain
#pragma unroll
        for (int t = 0; t < 4; ++t)
            av[t] = load_a_frag_f16(vv + (size_t)(t << 4) * NK + jn, NK, lane, 0);
    }

    // ---- epilogue: normalize (per-lane 1/l) and store f16 [o][p]
    const float inv = 1.f / l;
    _Float16* ob = o_m + (size_t)b * HID * NQ + (size_t)h * DHEAD * NQ + p0 + i;
#pragma unroll
    for (int t = 0; t < 4; ++t) {
#pragma unroll
        for (int r = 0; r < 8; ++r) {
            const int d = (t << 4) + r + (half << 3);
            ob[(size_t)d * NQ] = (_Float16)(acc[t][r] * inv);
        }
    }
}

// ---------------------------------------------------------------------------
extern "C" void kernel_launch(void* const* d_in, const int* in_sizes, int n_in,
                              void* d_out, int out_size, void* d_ws, size_t ws_size,
                              hipStream_t stream) {
    const float* x   = (const float*)d_in[0];   // [4,256,64,64]
    const float* ctx = (const float*)d_in[1];   // [4,512,32,32]
    const float* Wq  = (const float*)d_in[2];   // [512,256]
    const float* Wk  = (const float*)d_in[3];   // [512,512]
    const float* Wv  = (const float*)d_in[4];   // [512,512]
    const float* Wo  = (const float*)d_in[5];   // [256,512]
    const float* bo  = (const float*)d_in[6];   // [256]
    float* out = (float*)d_out;                 // [4,256,64,64]

    _Float16* ws  = (_Float16*)d_ws;
    _Float16* q_t = ws;                                     // [4*8][4096][64]
    _Float16* k_t = q_t + (size_t)NB * HEADS * NQ * DHEAD;  // [4*8][1024][64]
    _Float16* v_m = k_t + (size_t)NB * HEADS * NK * DHEAD;  // [4*8][64][1024]
    _Float16* o_m = v_m + (size_t)NB * HEADS * DHEAD * NK;  // [4][512][4096]

    const dim3 blk(256);
    // Q = Wq*x       : M=512 N=4096 K=256 -> 8192 tiles -> 1024 blocks/batch
    proj_kernel<float><<<dim3(1024, NB), blk, 0, stream>>>(
        Wq, x, q_t, nullptr, nullptr, HID, NQ, QDIM, 1);
    // K = Wk*ctx     : M=512 N=1024 K=512 -> 2048 tiles -> 256 blocks/batch
    proj_kernel<float><<<dim3(256, NB), blk, 0, stream>>>(
        Wk, ctx, k_t, nullptr, nullptr, HID, NK, CDIM, 1);
    // V = Wv*ctx
    proj_kernel<float><<<dim3(256, NB), blk, 0, stream>>>(
        Wv, ctx, v_m, nullptr, nullptr, HID, NK, CDIM, 0);
    // flash attention: 4096 queries / (8 waves * 16 cols) = 32 blocks per (b,h)
    attn_kernel<<<dim3(32, HEADS, NB), blk, 0, stream>>>(q_t, k_t, v_m, o_m);
    // final O projection + bias : M=256 N=4096 K=512 -> 4096 tiles
    proj_kernel<_Float16><<<dim3(512, NB), blk, 0, stream>>>(
        Wo, o_m, nullptr, out, bo, QDIM, NQ, HID, 2);
}
